// CktGNN_7765300871412
// MI455X (gfx1250) — matrix operand order repaired and checked by reading, not compiled, not run on hardware
//
#include <hip/hip_runtime.h>
#include <cmath>

#define B_TOT   4096
#define MAX_N   32
#define NVT     26
#define MAXPOS  9
#define HS      301
#define EMB     16
#define FEAT    8
#define NZ      56

#define KP      320            // padded K (301 -> 320, multiple of 32)
#define N1      912            // padded 903 (gh = Hin @ W_hh^T)
#define N2      608            // [Wg'(304) | Wm'(304)]
#define NTOT    (N1 + N2)      // 1520 packed weight columns
#define GSTRIDE 304            // padded g row stride
#define BT      16             // batch rows per workgroup (one WMMA M-tile)
#define NW      8              // waves per workgroup (wave32)
#define TPB     256

typedef __attribute__((ext_vector_type(8)))  _Float16 f16x8;
typedef __attribute__((ext_vector_type(16))) _Float16 v16h;
typedef __attribute__((ext_vector_type(8)))  float    v8f;

__device__ __forceinline__ float sigmoidf_(float x) {
    return 1.0f / (1.0f + __expf(-x));
}

// A fragment (16x32 f16) from LDS, A row-major [16][KP].
// Lanes 0-15: M=lane, VGPR0-3 K=0..7, VGPR4-7 K=16..23.
// Lanes 16-31: M=lane-16, VGPR0-3 K=8..15, VGPR4-7 K=24..31.
__device__ __forceinline__ v16h load_a_frag(const _Float16* A, int lane, int kt) {
    const int m = lane & 15, hg = lane >> 4;
    const _Float16* p = A + m * KP + kt * 32 + 8 * hg;
    f16x8 lo = *(const f16x8*)(p);        // K = 8*hg .. 8*hg+7
    f16x8 hi = *(const f16x8*)(p + 16);   // K = 16+8*hg .. 16+8*hg+7
    return __builtin_shufflevector(lo, hi, 0,1,2,3,4,5,6,7,8,9,10,11,12,13,14,15);
}

// B fragment (32x16 f16) from global, weights column-major WBcol[n][KP].
// Lanes 0-15 hold K=0..7,16..23 for N=lane; lanes 16-31 hold K=8..15,24..31 for N=lane-16.
__device__ __forceinline__ v16h load_b_frag(const _Float16* WBcol, int lane, int kt) {
    const int n = lane & 15, kg = lane >> 4;
    const _Float16* p = WBcol + n * KP + kt * 32 + kg * 8;
    __builtin_prefetch(p + 32, 0, 1);     // next K-chunk -> global_prefetch_b8
    f16x8 lo = *(const f16x8*)(p);
    f16x8 hi = *(const f16x8*)(p + 16);
    return __builtin_shufflevector(lo, hi, 0,1,2,3,4,5,6,7,8,9,10,11,12,13,14,15);
}

// Pack [W_hh^T | Wg'(h-part) | Wm'(h-part)] into column-major f16, K padded to 320.
__global__ void pack_weights(const float* __restrict__ Wg, const float* __restrict__ Wm,
                             const float* __restrict__ Whh, _Float16* __restrict__ WB) {
    int idx = blockIdx.x * blockDim.x + threadIdx.x;
    if (idx >= NTOT * KP) return;
    int n = idx / KP, k = idx - n * KP;
    float v = 0.f;
    if (n < N1) {                                  // gh columns: W_hh^T[k][j] = W_hh[j][k]
        if (n < 3 * HS && k < HS) v = Whh[(size_t)n * HS + k];
    } else {
        int c = n - N1;
        if (c < GSTRIDE) {                         // Wg h-part, column j = c
            if (c < HS && k < HS) v = Wg[(size_t)k * HS + c];
        } else {                                   // Wm h-part, column j = c-304
            int j = c - GSTRIDE;
            if (j < HS && k < HS) v = Wm[(size_t)k * HS + j];
        }
    }
    WB[idx] = (_Float16)v;
}

__global__ __launch_bounds__(TPB) void ckt_main(
    const int* __restrict__ types, const int* __restrict__ pos,
    const int* __restrict__ adj,   const float* __restrict__ hdf,
    const float* __restrict__ Wg,  const float* __restrict__ bg,
    const float* __restrict__ Wm,  const float* __restrict__ W_ih,
    const float* __restrict__ b_ih,const float* __restrict__ b_hh,
    const float* __restrict__ Wd1, const float* __restrict__ bd1,
    const float* __restrict__ Wd2, const float* __restrict__ bd2,
    const float* __restrict__ Wmu, const float* __restrict__ bmu,
    const float* __restrict__ Wlv, const float* __restrict__ blv,
    const _Float16* __restrict__ WB, _Float16* __restrict__ g_ws,
    float* __restrict__ out)
{
    __shared__ __align__(16) _Float16 A_lds[BT * KP];      // 10 KB  f16 A tile (Hin or h)
    __shared__ float scratch[BT * N1];                      // 58 KB  GEMM output staging
    __shared__ float hin_lds[BT * GSTRIDE];                 // 19 KB
    __shared__ float h_lds[BT * GSTRIDE];                   // 19 KB  (h_v; final = Hg)
    __shared__ int   adj_lds[BT * MAX_N];
    __shared__ int   type_lds[BT];
    __shared__ int   pos_lds[BT];
    __shared__ float hd_lds[BT * FEAT];

    const int tid  = threadIdx.x;
    const int wave = tid >> 5;
    const int lane = tid & 31;
    const int b0   = blockIdx.x * BT;

    for (int v = 0; v < MAX_N; ++v) {
        // ---- phase 0: stage adj row v, type/pos for this step ----
        for (int idx = tid; idx < BT * MAX_N; idx += TPB) {
            int m = idx >> 5, n = idx & 31;
            adj_lds[idx] = adj[((size_t)(b0 + m) * MAX_N + v) * MAX_N + n];
        }
        if (tid < BT) {
            type_lds[tid] = types[(size_t)(b0 + tid) * MAX_N + v];
            pos_lds[tid]  = pos[(size_t)(b0 + tid) * MAX_N + v];
        }
        __syncthreads();

        // ---- phase 1: Hin[m,k] = sum_{n<v} adj[m,v,n] * g[m,n,k]; fill f16 A tile ----
        for (int idx = tid; idx < BT * KP; idx += TPB) {
            int m = idx / KP, k = idx - m * KP;
            float acc = 0.f;
            if (k < HS && v > 0) {
                const _Float16* gb = g_ws + (size_t)(b0 + m) * MAX_N * GSTRIDE + k;
                const int* am = adj_lds + m * MAX_N;
                for (int n = 0; n < v; ++n)
                    if (am[n]) acc += (float)gb[(size_t)n * GSTRIDE];
            }
            if (k < GSTRIDE) hin_lds[m * GSTRIDE + k] = acc;
            A_lds[idx] = (_Float16)acc;              // zero in K-pad region
        }
        __syncthreads();

        // ---- phase 2: GEMM1  gh = Hin @ W_hh^T  (16 x 912), f16 WMMA / f32 acc ----
        for (int tile = wave; tile < N1 / 16; tile += NW) {
            const _Float16* bcol = WB + (size_t)(tile * 16) * KP;
            v8f c = {};
            for (int kt = 0; kt < KP / 32; ++kt) {
                v16h a = load_a_frag(A_lds, lane, kt);
                v16h b = load_b_frag(bcol, lane, kt);
                c = __builtin_amdgcn_wmma_f32_16x16x32_f16(false, a, false, b,
                                                           (short)0, c, false, false);
            }
            int n0 = tile * 16 + (lane & 15);
            int mr = (lane >> 4) * 8;                // C layout: VGPR i -> M = i + 8*(lane>=16)
            #pragma unroll
            for (int i = 0; i < 8; ++i) scratch[(mr + i) * N1 + n0] = c[i];
        }
        __syncthreads();

        // ---- phase 3: GRU elementwise; gi via one-hot gathers from W_ih ----
        for (int idx = tid; idx < BT * HS; idx += TPB) {
            int m = idx / HS, j = idx - m * HS;
            int t = type_lds[m], p = pos_lds[m];
            float gir = W_ih[(size_t)j * 35 + t]            + W_ih[(size_t)j * 35 + NVT + p]            + b_ih[j];
            float giz = W_ih[(size_t)(HS + j) * 35 + t]     + W_ih[(size_t)(HS + j) * 35 + NVT + p]     + b_ih[HS + j];
            float gin = W_ih[(size_t)(2*HS + j) * 35 + t]   + W_ih[(size_t)(2*HS + j) * 35 + NVT + p]   + b_ih[2*HS + j];
            float ghr = scratch[m * N1 + j]          + b_hh[j];
            float ghz = scratch[m * N1 + HS + j]     + b_hh[HS + j];
            float ghn = scratch[m * N1 + 2*HS + j]   + b_hh[2*HS + j];
            float r  = sigmoidf_(gir + ghr);
            float z  = sigmoidf_(giz + ghz);
            float nn = tanhf(gin + r * ghn);
            float hv = (1.f - z) * nn + z * hin_lds[m * GSTRIDE + j];
            h_lds[m * GSTRIDE + j] = hv;
            A_lds[m * KP + j] = (_Float16)hv;        // pad cols stay zero from phase 1
        }
        __syncthreads();

        // ---- phase 4: GEMM2  [gpre | mraw] = h_v @ [Wg' | Wm']  (16 x 608) ----
        for (int tile = wave; tile < N2 / 16; tile += NW) {
            const _Float16* bcol = WB + (size_t)(N1 + tile * 16) * KP;
            v8f c = {};
            for (int kt = 0; kt < KP / 32; ++kt) {
                v16h a = load_a_frag(A_lds, lane, kt);
                v16h b = load_b_frag(bcol, lane, kt);
                c = __builtin_amdgcn_wmma_f32_16x16x32_f16(false, a, false, b,
                                                           (short)0, c, false, false);
            }
            int n0 = tile * 16 + (lane & 15);
            int mr = (lane >> 4) * 8;
            #pragma unroll
            for (int i = 0; i < 8; ++i) scratch[(mr + i) * N2 + n0] = c[i];
        }
        __syncthreads();

        // ---- phase 5: g_v = sigmoid(gpre + Wg[301+p] + bg) * (mraw + Wm[301+p]) ----
        for (int idx = tid; idx < BT * HS; idx += TPB) {
            int m = idx / HS, j = idx - m * HS;
            int p = pos_lds[m];
            float gpre = scratch[m * N2 + j]           + bg[j] + Wg[(size_t)(HS + p) * HS + j];
            float mr   = scratch[m * N2 + GSTRIDE + j]         + Wm[(size_t)(HS + p) * HS + j];
            float gv = sigmoidf_(gpre) * mr;
            g_ws[((size_t)(b0 + m) * MAX_N + v) * GSTRIDE + j] = (_Float16)gv;
        }
        __threadfence();    // make g_ws stores visible before next step's Hin gather
        __syncthreads();
    }

    // ---- Hd head: relu(hdf @ Wd1 + bd1) @ Wd2 + bd2  (tiny, VALU) ----
    if (tid < BT * FEAT) {
        int m = tid >> 3, f = tid & 7;
        const float* hb = hdf + (size_t)(b0 + m) * (3 * MAXPOS);
        float acc = bd2[f];
        for (int i = 0; i < EMB; ++i) {
            float e = bd1[i];
            for (int c = 0; c < 3 * MAXPOS; ++c) e += hb[c] * Wd1[c * EMB + i];
            e = fmaxf(e, 0.f);
            acc += e * Wd2[i * FEAT + f];
        }
        hd_lds[tid] = acc;
    }
    __syncthreads();

    // ---- output: mu / logvar = [Hg | Hd] @ {Wmu,Wlv} + {bmu,blv} ----
    for (int idx = tid; idx < BT * NZ * 2; idx += TPB) {
        int m = idx / (2 * NZ), q = idx - m * (2 * NZ);
        int which = q / NZ, o = q - which * NZ;
        const float* W  = which ? Wlv : Wmu;
        const float* bb = which ? blv : bmu;
        float acc = bb[o];
        const float* hm = h_lds + m * GSTRIDE;       // h at v=31 == Hg
        for (int k = 0; k < HS; ++k)   acc += hm[k] * W[(size_t)k * NZ + o];
        for (int f = 0; f < FEAT; ++f) acc += hd_lds[m * FEAT + f] * W[(size_t)(HS + f) * NZ + o];
        out[(size_t)which * B_TOT * NZ + (size_t)(b0 + m) * NZ + o] = acc;
    }
}

extern "C" void kernel_launch(void* const* d_in, const int* in_sizes, int n_in,
                              void* d_out, int out_size, void* d_ws, size_t ws_size,
                              hipStream_t stream) {
    const int*   types = (const int*)  d_in[0];
    const int*   pos   = (const int*)  d_in[1];
    const int*   adj   = (const int*)  d_in[2];
    const float* hdf   = (const float*)d_in[3];
    const float* Wg    = (const float*)d_in[4];
    const float* bg    = (const float*)d_in[5];
    const float* Wm    = (const float*)d_in[6];
    const float* W_ih  = (const float*)d_in[7];
    const float* W_hh  = (const float*)d_in[8];
    const float* b_ih  = (const float*)d_in[9];
    const float* b_hh  = (const float*)d_in[10];
    const float* Wd1   = (const float*)d_in[11];
    const float* bd1   = (const float*)d_in[12];
    const float* Wd2   = (const float*)d_in[13];
    const float* bd2   = (const float*)d_in[14];
    const float* Wmu   = (const float*)d_in[15];
    const float* bmu   = (const float*)d_in[16];
    const float* Wlv   = (const float*)d_in[17];
    const float* blv   = (const float*)d_in[18];
    float* out = (float*)d_out;

    // ws layout: [ packed f16 weights: 1520*320*2 = 972800 B | g buffer: 4096*32*304*2 ~= 76 MB ]
    _Float16* WB   = (_Float16*)d_ws;
    _Float16* g_ws = (_Float16*)((char*)d_ws + (size_t)NTOT * KP * sizeof(_Float16));

    pack_weights<<<(NTOT * KP + TPB - 1) / TPB, TPB, 0, stream>>>(Wg, Wm, W_hh, WB);
    ckt_main<<<B_TOT / BT, TPB, 0, stream>>>(types, pos, adj, hdf, Wg, bg, Wm, W_ih,
                                             b_ih, b_hh, Wd1, bd1, Wd2, bd2,
                                             Wmu, bmu, Wlv, blv, WB, g_ws, out);
}
